// Graph_module_net_0_loss_18631568130104
// MI455X (gfx1250) — matrix operand order repaired and checked by reading, not compiled, not run on hardware
//
#include <hip/hip_runtime.h>
#include <hip/hip_bf16.h>
#include <stdint.h>

// ---------------------------------------------------------------------------
// MI455X (gfx1250, wave32): WMMA f16 GEMMs + TDM (tensor_load_to_lds) staging.
// ---------------------------------------------------------------------------

typedef __attribute__((ext_vector_type(16))) _Float16 v16h;
typedef __attribute__((ext_vector_type(8)))  float    v8f;
typedef __attribute__((ext_vector_type(4)))  unsigned int v4u;
typedef __attribute__((ext_vector_type(8)))  int v8i;
typedef __attribute__((ext_vector_type(4)))  int v4i;

#define NB    4
#define NUM   1024
#define CCH   256
#define NH    4
#define DK    64
#define NGRP  4
#define SFLT  (NB * NUM * CCH)          // 1,048,576 elements
#define NNEL  (NB * NH * NUM * NUM)     // 16,777,216 elements

#define WMMA16(acc, a, b)                                                     \
  (acc) = __builtin_amdgcn_wmma_f32_16x16x32_f16(false, (a), false, (b),      \
                                                 (short)0, (acc), false, false)

// --- WMMA fragment loaders (contiguous k = 16h + s, same order in A and B) ----------------------
__device__ __forceinline__ v16h load_a_frag(const _Float16* A, int lda) {
  const int lane = threadIdx.x & 31;
  const int m  = lane & 15;
  const int kb = (lane & 16) ? 16 : 0;
  const _Float16* r = A + (size_t)m * lda;
  v16h a;
#pragma unroll
  for (int v = 0; v < 4; ++v) {
    a[2 * v]         = r[kb + 2 * v];
    a[2 * v + 1]     = r[kb + 2 * v + 1];
    a[8 + 2 * v]     = r[kb + 8 + 2 * v];
    a[8 + 2 * v + 1] = r[kb + 8 + 2 * v + 1];
  }
  return a;
}

// B stored row-major K x N (ldb = row stride).
__device__ __forceinline__ v16h load_b_rm(const _Float16* B, int ldb) {
  const int lane = threadIdx.x & 31;
  const int n  = lane & 15;
  const int kb = (lane & 16) ? 16 : 0;
  v16h b;
#pragma unroll
  for (int v = 0; v < 8; ++v) {
    b[2 * v]     = B[(size_t)(kb + 2 * v) * ldb + n];
    b[2 * v + 1] = B[(size_t)(kb + 2 * v + 1) * ldb + n];
  }
  return b;
}

// B given as N x K row-major (i.e. col-major K x N).
__device__ __forceinline__ v16h load_b_cm(const _Float16* Bt, int ldb) {
  const int lane = threadIdx.x & 31;
  const int n  = lane & 15;
  const int kb = (lane & 16) ? 16 : 0;
  const _Float16* r = Bt + (size_t)n * ldb;
  v16h b;
#pragma unroll
  for (int v = 0; v < 8; ++v) {
    b[2 * v]     = r[kb + 2 * v];
    b[2 * v + 1] = r[kb + 2 * v + 1];
  }
  return b;
}

// --- TDM: 2D tensor tile -> LDS (ISA ch.8 D# layout, data_size = 2 bytes) ---
__device__ __forceinline__ void tdm_load_2d_f16(uint32_t lds_off, const void* gaddr,
                                                uint32_t tile_d0, uint32_t tile_d1,
                                                uint32_t tensor_d0, uint32_t tensor_d1,
                                                uint64_t stride0_elems) {
  const uint64_t ga = (uint64_t)gaddr;
  v4u g0;
  g0[0] = 1u;                                               // count=1 (valid), user mode
  g0[1] = lds_off;                                          // lds_addr (bytes)
  g0[2] = (uint32_t)ga;                                     // global_addr[31:0]
  g0[3] = (uint32_t)((ga >> 32) & 0x01FFFFFFu) | (2u << 30);// addr[56:32] | type=2
  uint32_t d0 = (1u << 16);                                 // data_size = 1 -> 2 bytes
  uint32_t d1 = (tensor_d0 & 0xFFFFu) << 16;                // tensor_dim0[15:0] @ bit48
  uint32_t d2 = ((tensor_d0 >> 16) & 0xFFFFu)               // tensor_dim0[31:16]
              | ((tensor_d1 & 0xFFFFu) << 16);              // tensor_dim1[15:0] @ bit80
  uint32_t d3 = ((tensor_d1 >> 16) & 0xFFFFu)               // tensor_dim1[31:16]
              | ((tile_d0 & 0xFFFFu) << 16);                // tile_dim0 @ bit112
  uint32_t d4 = (tile_d1 & 0xFFFFu);                        // tile_dim1 @ bit128
  uint32_t d5 = (uint32_t)(stride0_elems & 0xFFFFFFFFu);    // dim0_stride @ bit160
  uint32_t d6 = (uint32_t)((stride0_elems >> 32) & 0xFFFFu);
  v8i g1;
  g1[0] = (int)d0; g1[1] = (int)d1; g1[2] = (int)d2; g1[3] = (int)d3;
  g1[4] = (int)d4; g1[5] = (int)d5; g1[6] = (int)d6; g1[7] = 0;
  v4i z4 = {0, 0, 0, 0};
#if defined(__clang_major__) && (__clang_major__ >= 23)
  v8i z8 = {0, 0, 0, 0, 0, 0, 0, 0};
  __builtin_amdgcn_tensor_load_to_lds(g0, g1, z4, z4, z8, 0);
#else
  __builtin_amdgcn_tensor_load_to_lds(g0, g1, z4, z4, 0);
#endif
}

// ---------------------------------------------------------------------------
__global__ void f2h_kernel(const float* __restrict__ src, _Float16* __restrict__ dst, int n) {
  int t = blockIdx.x * blockDim.x + threadIdx.x;
  if (t < n) dst[t] = (_Float16)src[t];
}

__global__ void zero_u32_kernel(unsigned int* p, int n) {
  int t = blockIdx.x * blockDim.x + threadIdx.x;
  if (t < n) p[t] = 0u;
}

// per-head L2 norms
__global__ void norms_kernel(const _Float16* __restrict__ x, float* __restrict__ nrm) {
  int t = blockIdx.x * blockDim.x + threadIdx.x;
  if (t >= NB * NUM * NH) return;
  int hh = t & 3;
  int bn = t >> 2;
  const _Float16* p = x + (size_t)bn * CCH + hh * DK;
  float s = 0.f;
#pragma unroll 8
  for (int d = 0; d < DK; ++d) {
    float v = (float)p[d];
    s += v * v;
  }
  nrm[t] = sqrtf(s);
}

// cosine-sim dots via WMMA: attn[bh][i][j] = relu(dot * rcp(denom))
__global__ void attn_dots_kernel(const _Float16* __restrict__ x,
                                 const float* __restrict__ nrm,
                                 float* __restrict__ attn) {
  const int wave = threadIdx.x >> 5;
  const int i0 = blockIdx.x * 16;
  const int j0 = (blockIdx.y * 4 + wave) * 16;
  const int bh = blockIdx.z;
  const int b = bh >> 2, hh = bh & 3;

  const _Float16* A  = x + ((size_t)(b * NUM + i0)) * CCH + hh * DK;
  const _Float16* Bt = x + ((size_t)(b * NUM + j0)) * CCH + hh * DK;

  v8f acc = {};
#pragma unroll
  for (int k = 0; k < DK; k += 32) {
    v16h af = load_a_frag(A + k, CCH);
    v16h bf = load_b_cm(Bt + k, CCH);
    WMMA16(acc, af, bf);
  }

  const int lane = threadIdx.x & 31;
  const int jc = j0 + (lane & 15);
  const int mb = (lane & 16) ? 8 : 0;
  const float nj = nrm[(size_t)(b * NUM + jc) * NH + hh];
#pragma unroll
  for (int v = 0; v < 8; ++v) {
    const int m = i0 + mb + v;
    const float ni = nrm[(size_t)(b * NUM + m) * NH + hh];
    const float rden = __builtin_amdgcn_rcpf(fmaxf(ni * nj, 1e-8f));  // 1x v_rcp_f32
    attn[((size_t)bh * NUM + m) * NUM + jc] = fmaxf(acc[v] * rden, 0.f);
  }
}

// global top-4 per row -> union column mask.  One wave per (bh,i) row.
__global__ void topk_mask_kernel(const float* __restrict__ attn,
                                 unsigned int* __restrict__ colmask) {
  __shared__ float svals[8][128];
  __shared__ int   sidx[8][128];
  const int wave = threadIdx.x >> 5, lane = threadIdx.x & 31;
  const int row = blockIdx.x * 8 + wave;
  const float* a = attn + (size_t)row * NUM;

  float tv[4] = {-1.f, -1.f, -1.f, -1.f};
  int   ti[4] = {0, 0, 0, 0};
  for (int j = lane; j < NUM; j += 32) {
    float v = a[j];
    if (v > tv[3]) {
      int p = 3;
      while (p > 0 && v > tv[p - 1]) { tv[p] = tv[p - 1]; ti[p] = ti[p - 1]; --p; }
      tv[p] = v; ti[p] = j;
    }
  }
#pragma unroll
  for (int q = 0; q < 4; ++q) {
    svals[wave][lane * 4 + q] = tv[q];
    sidx[wave][lane * 4 + q]  = ti[q];
  }
  __syncthreads();
  if (lane == 0) {
    float bv[4] = {-1.f, -1.f, -1.f, -1.f};
    int   bi[4] = {0, 0, 0, 0};
    for (int q = 0; q < 128; ++q) {
      float v = svals[wave][q];
      int   j = sidx[wave][q];
      if (v > bv[3]) {
        int p = 3;
        while (p > 0 && v > bv[p - 1]) { bv[p] = bv[p - 1]; bi[p] = bi[p - 1]; --p; }
        bv[p] = v; bi[p] = j;
      }
    }
#pragma unroll
    for (int q = 0; q < 4; ++q) atomicOr(&colmask[bi[q]], 1u);
  }
}

// P[bh][j][i] = f16((attn[bh][i][j]*roi*vec + diag)/4)
__global__ void mask_pack_kernel(const float* __restrict__ attn,
                                 const int* __restrict__ roi,
                                 const int* __restrict__ sm,
                                 const unsigned int* __restrict__ colmask,
                                 _Float16* __restrict__ P) {
  size_t t = (size_t)blockIdx.x * blockDim.x + threadIdx.x;  // ((bh*NUM+j)*NUM+i)
  if (t >= (size_t)NNEL) return;
  const int i  = (int)(t & (NUM - 1));
  const int j  = (int)((t >> 10) & (NUM - 1));
  const int bh = (int)(t >> 20);
  const int b  = bh >> 2;
  const float a = attn[((size_t)bh * NUM + i) * NUM + j];
  const float r = (float)(roi[((size_t)b * NUM + i) * NUM + j] * sm[b * NUM + j]);
  const float m = colmask[j] ? 1.f : 0.f;
  const float f = (i == j && sm[b * NUM + i] == 0) ? 1.f : 0.f;
  P[t] = (_Float16)((a * r * m + f) * 0.25f);
}

// Grouped 1x1 conv + ReLU via WMMA.  grid (NUM/16, NB*NGRP), block 128.
__global__ void gconv_kernel(const _Float16* __restrict__ x,
                             const _Float16* __restrict__ wh,
                             const float* __restrict__ bias,
                             float* __restrict__ outf,
                             _Float16* __restrict__ outh) {
  const int wave = threadIdx.x >> 5;
  const int i0 = blockIdx.x * 16;
  const int bg = blockIdx.y;
  const int b = bg >> 2, g = bg & 3;

  const _Float16* A = x + ((size_t)(b * NUM + i0)) * CCH + g * 64;
  const _Float16* W = wh + (size_t)g * 4096 + (wave * 16) * 64;

  v8f acc = {};
#pragma unroll
  for (int k = 0; k < 64; k += 32) {
    v16h af = load_a_frag(A + k, CCH);
    v16h bf = load_b_cm(W + k, 64);
    WMMA16(acc, af, bf);
  }

  const int lane = threadIdx.x & 31;
  const int o  = wave * 16 + (lane & 15);
  const int mb = (lane & 16) ? 8 : 0;
  const int ch = g * 64 + o;
  const float bv = bias[ch];
#pragma unroll
  for (int v = 0; v < 8; ++v) {
    const int n = i0 + mb + v;
    const float val = fmaxf(acc[v] + bv, 0.f);
    const size_t off = ((size_t)b * CCH + ch) * NUM + n;
    outf[off] = val;
    outh[off] = (_Float16)val;
  }
}

// Apply attention: o[bh][c][i] = sum_j outh[bh][c][j] * P[bh][j][i]   (K=1024)
// TDM double-buffered LDS staging of 64x64 A and P tiles; WMMA fed from LDS.
// grid (NUM/64, NB*NH), block 128 (4 waves over i-tiles; each wave all 4 c-tiles)
#define KCH 64
template <int MODE>  // 0 = stage1 (add + f16 out), 1 = stage2 (raw f32 out)
__global__ void apply_kernel(const _Float16* __restrict__ outh,
                             const _Float16* __restrict__ P,
                             const float* __restrict__ outf,
                             _Float16* __restrict__ x2h,
                             float* __restrict__ y) {
  __shared__ _Float16 As[2][64 * KCH] __attribute__((aligned(16)));  // [c][k]
  __shared__ _Float16 Ps[2][KCH * 64] __attribute__((aligned(16)));  // [k][i]

  const int wave = threadIdx.x >> 5;
  const int iblk = blockIdx.x * 64;              // block's 64 i-columns
  const int bh = blockIdx.y;
  const int b = bh >> 2, hh = bh & 3;

  const _Float16* Ag = outh + ((size_t)b * CCH + hh * 64) * NUM;   // 64 x NUM, ld NUM
  const _Float16* Pg = P + (size_t)bh * NUM * NUM + iblk;          // rows k, ld NUM

  // chunk ch covers K = [ch*KCH, ch*KCH+KCH)
  const int NCHUNK = NUM / KCH;                  // 16
  if (wave == 0) {
    tdm_load_2d_f16((uint32_t)(size_t)&As[0][0], Ag, KCH, 64, NUM, 64, NUM);
    tdm_load_2d_f16((uint32_t)(size_t)&Ps[0][0], Pg, 64, KCH, NUM - iblk, NUM, NUM);
  }

  v8f acc[4] = {v8f{}, v8f{}, v8f{}, v8f{}};
  for (int ch = 0; ch < NCHUNK; ++ch) {
    if (wave == 0) {
      if (ch + 1 < NCHUNK) {
        const int nb = (ch + 1) & 1;
        tdm_load_2d_f16((uint32_t)(size_t)&As[nb][0], Ag + (ch + 1) * KCH,
                        KCH, 64, NUM, 64, NUM);
        tdm_load_2d_f16((uint32_t)(size_t)&Ps[nb][0], Pg + (size_t)(ch + 1) * KCH * NUM,
                        64, KCH, NUM - iblk, NUM, NUM);
        __builtin_amdgcn_s_wait_tensorcnt(2);    // chunk ch landed, ch+1 in flight
      } else {
        __builtin_amdgcn_s_wait_tensorcnt(0);
      }
    }
    __syncthreads();                             // publish LDS tiles
    const _Float16* Al = &As[ch & 1][0];
    const _Float16* Pl = &Ps[ch & 1][0];
#pragma unroll
    for (int kl = 0; kl < KCH; kl += 32) {
      v16h bf = load_b_rm(Pl + (size_t)kl * 64 + wave * 16, 64);   // ds loads
#pragma unroll
      for (int ct = 0; ct < 4; ++ct) {
        v16h af = load_a_frag(Al + (size_t)(ct * 16) * KCH + kl, KCH);
        WMMA16(acc[ct], af, bf);
      }
    }
    __syncthreads();                             // tile consumed, buffer reusable
  }

  const int lane = threadIdx.x & 31;
  const int ic = iblk + wave * 16 + (lane & 15);
  const int mb = (lane & 16) ? 8 : 0;
#pragma unroll
  for (int ct = 0; ct < 4; ++ct) {
#pragma unroll
    for (int v = 0; v < 8; ++v) {
      const int c = ct * 16 + mb + v;
      const size_t dsto = ((size_t)b * NUM + ic) * CCH + hh * 64 + c;
      if (MODE == 0) {
        const float val = outf[((size_t)b * CCH + hh * 64 + c) * NUM + ic] + acc[ct][v];
        x2h[dsto] = (_Float16)val;
      } else {
        y[dsto] = acc[ct][v];
      }
    }
  }
}

// gts = relu(gt_feat @ gt_w^T + gt_b) : M=4096, N=256, K=256
__global__ void gts_kernel(const _Float16* __restrict__ A,
                           const _Float16* __restrict__ Wt,
                           const float* __restrict__ bias,
                           float* __restrict__ out) {
  const int wave = threadIdx.x >> 5;
  const int m0 = blockIdx.x * 16;
  const int o0 = (blockIdx.y * 4 + wave) * 16;

  v8f acc = {};
#pragma unroll
  for (int k = 0; k < 256; k += 32) {
    v16h af = load_a_frag(A + (size_t)m0 * 256 + k, 256);
    v16h bf = load_b_cm(Wt + (size_t)o0 * 256 + k, 256);
    WMMA16(acc, af, bf);
  }

  const int lane = threadIdx.x & 31;
  const int o  = o0 + (lane & 15);
  const int mb = (lane & 16) ? 8 : 0;
  const float bv = bias[o];
#pragma unroll
  for (int v = 0; v < 8; ++v) {
    const int m = m0 + mb + v;
    out[(size_t)m * 256 + o] = fmaxf(acc[v] + bv, 0.f);
  }
}

// LayerNorm + final assembly.  One 256-thread block per (b,i).
__global__ void ln_final_kernel(const float* __restrict__ y,
                                const float* __restrict__ out2f,
                                const float* __restrict__ lnw,
                                const float* __restrict__ lnb,
                                float* __restrict__ out0,
                                float* __restrict__ nf) {
  const int bi = blockIdx.x;
  const int c  = threadIdx.x;
  const int b  = bi >> 10, i = bi & (NUM - 1);

  __shared__ float red[256];
  __shared__ float mu_s, var_s;

  const float v = y[(size_t)bi * CCH + c];
  red[c] = v;
  __syncthreads();
  for (int s = 128; s > 0; s >>= 1) {
    if (c < s) red[c] += red[c + s];
    __syncthreads();
  }
  if (c == 0) mu_s = red[0] * (1.f / CCH);
  __syncthreads();
  const float mu = mu_s;
  const float d = v - mu;
  red[c] = d * d;
  __syncthreads();
  for (int s = 128; s > 0; s >>= 1) {
    if (c < s) red[c] += red[c + s];
    __syncthreads();
  }
  if (c == 0) var_s = red[0] * (1.f / CCH);
  __syncthreads();
  const float rstd = rsqrtf(var_s + 1e-6f);
  const float nfv = d * rstd * lnw[c] + lnb[c];
  nf[(size_t)bi * CCH + c]   = nfv;
  out0[(size_t)bi * CCH + c] = out2f[((size_t)b * CCH + c) * NUM + i] + nfv;
}

// ---------------------------------------------------------------------------
extern "C" void kernel_launch(void* const* d_in, const int* in_sizes, int n_in,
                              void* d_out, int out_size, void* d_ws, size_t ws_size,
                              hipStream_t stream) {
  const float* in_x   = (const float*)d_in[0];
  const int*   roi    = (const int*)  d_in[1];
  const int*   sm     = (const int*)  d_in[2];
  const float* gtf    = (const float*)d_in[3];
  const float* w1     = (const float*)d_in[4];
  const float* b1     = (const float*)d_in[5];
  const float* w2     = (const float*)d_in[6];
  const float* b2     = (const float*)d_in[7];
  const float* gtw    = (const float*)d_in[8];
  const float* gtb    = (const float*)d_in[9];
  const float* lnw    = (const float*)d_in[10];
  const float* lnb    = (const float*)d_in[11];

  float* out0 = (float*)d_out;
  float* gts  = out0 + SFLT;
  float* nf   = out0 + 2 * (size_t)SFLT;

  char* ws = (char*)d_ws;
  size_t off = 0;
  auto alloc = [&](size_t bytes) {
    void* p = ws + off;
    off = (off + bytes + 255) & ~(size_t)255;
    return p;
  };
  _Float16* xh16  = (_Float16*)alloc((size_t)SFLT * 2);
  _Float16* gth16 = (_Float16*)alloc((size_t)SFLT * 2);
  _Float16* gtwh  = (_Float16*)alloc((size_t)256 * 256 * 2);
  _Float16* w1h   = (_Float16*)alloc((size_t)NGRP * 64 * 64 * 2);
  _Float16* w2h   = (_Float16*)alloc((size_t)NGRP * 64 * 64 * 2);
  float*    nrm1  = (float*)   alloc((size_t)NB * NUM * NH * 4);
  float*    nrm2  = (float*)   alloc((size_t)NB * NUM * NH * 4);
  float*    attn  = (float*)   alloc((size_t)NNEL * 4);
  _Float16* Pbuf  = (_Float16*)alloc((size_t)NNEL * 2);
  float*    convf = (float*)   alloc((size_t)SFLT * 4);
  _Float16* convh = (_Float16*)alloc((size_t)SFLT * 2);
  _Float16* x2h   = (_Float16*)alloc((size_t)SFLT * 2);
  float*    ybuf  = (float*)   alloc((size_t)SFLT * 4);
  unsigned* cmask = (unsigned*)alloc((size_t)NUM * 4);
  (void)ws_size; (void)in_sizes; (void)n_in; (void)out_size;

  const dim3 blk128(128), blk256(256);

  f2h_kernel<<<(SFLT + 255) / 256, blk256, 0, stream>>>(in_x, xh16, SFLT);
  f2h_kernel<<<(SFLT + 255) / 256, blk256, 0, stream>>>(gtf, gth16, SFLT);
  f2h_kernel<<<(256 * 256 + 255) / 256, blk256, 0, stream>>>(gtw, gtwh, 256 * 256);
  f2h_kernel<<<(NGRP * 64 * 64 + 255) / 256, blk256, 0, stream>>>(w1, w1h, NGRP * 64 * 64);
  f2h_kernel<<<(NGRP * 64 * 64 + 255) / 256, blk256, 0, stream>>>(w2, w2h, NGRP * 64 * 64);

  gts_kernel<<<dim3(NB * NUM / 16, 256 / 64), blk128, 0, stream>>>(gth16, gtwh, gtb, gts);

  // ================= stage 1 =================
  norms_kernel<<<(NB * NUM * NH + 255) / 256, blk256, 0, stream>>>(xh16, nrm1);
  attn_dots_kernel<<<dim3(NUM / 16, NUM / 64, NB * NH), blk128, 0, stream>>>(xh16, nrm1, attn);
  zero_u32_kernel<<<4, blk256, 0, stream>>>(cmask, NUM);
  topk_mask_kernel<<<(NB * NH * NUM) / 8, blk256, 0, stream>>>(attn, cmask);
  mask_pack_kernel<<<(unsigned)(NNEL / 256), blk256, 0, stream>>>(attn, roi, sm, cmask, Pbuf);
  gconv_kernel<<<dim3(NUM / 16, NB * NGRP), blk128, 0, stream>>>(xh16, w1h, b1, convf, convh);
  apply_kernel<0><<<dim3(NUM / 64, NB * NH), blk128, 0, stream>>>(convh, Pbuf, convf, x2h, nullptr);

  // ================= stage 2 =================
  norms_kernel<<<(NB * NUM * NH + 255) / 256, blk256, 0, stream>>>(x2h, nrm2);
  attn_dots_kernel<<<dim3(NUM / 16, NUM / 64, NB * NH), blk128, 0, stream>>>(x2h, nrm2, attn);
  zero_u32_kernel<<<4, blk256, 0, stream>>>(cmask, NUM);
  topk_mask_kernel<<<(NB * NH * NUM) / 8, blk256, 0, stream>>>(attn, cmask);
  mask_pack_kernel<<<(unsigned)(NNEL / 256), blk256, 0, stream>>>(attn, roi, sm, cmask, Pbuf);
  gconv_kernel<<<dim3(NUM / 16, NB * NGRP), blk128, 0, stream>>>(x2h, w2h, b2, convf, convh);
  apply_kernel<1><<<dim3(NUM / 64, NB * NH), blk128, 0, stream>>>(convh, Pbuf, convf, nullptr, ybuf);

  ln_final_kernel<<<NB * NUM, blk256, 0, stream>>>(ybuf, convf, lnw, lnb, out0, nf);
}
